// Network_32787780337671
// MI455X (gfx1250) — compile-verified, matrix operand order checked by reading, not run on hardware
//
#include <hip/hip_runtime.h>
#include <stdint.h>

typedef unsigned short u16;
typedef __attribute__((ext_vector_type(16))) __bf16 v16bf;
typedef __attribute__((ext_vector_type(8)))  float  v8f;

// ---------- helpers ----------
__device__ __forceinline__ u16 f2bf(float x) {
  union { float f; uint32_t u; } c; c.f = x;
  uint32_t u = c.u;
  uint32_t r = (u + 0x7FFFu + ((u >> 16) & 1u)) >> 16;
  return (u16)r;
}

union FragU { v16bf v; uint32_t u[8]; };

// Load a 16x32 bf16 operand fragment from a row-major (ld halves) tile.
// Per CDNA5 ISA 16-bit A/B layout: lane%16 = row (M or N), lanes>=16 take K+8,
// VGPR v holds K pair {base + (v&3)*2, +1}, VGPRs 4..7 add K+16.
__device__ __forceinline__ v16bf load_frag(const u16* base, int ld, int lane) {
  int r  = lane & 15;
  int kb = (lane & 16) ? 8 : 0;
  FragU f;
#pragma unroll
  for (int v = 0; v < 8; ++v) {
    int k = kb + ((v & 4) << 2) + ((v & 3) << 1);
    f.u[v] = *reinterpret_cast<const uint32_t*>(base + r * ld + k);
  }
  return f.v;
}

__device__ __forceinline__ v8f wmma_bf16(v16bf a, v16bf b, v8f c) {
  return __builtin_amdgcn_wmma_f32_16x16x32_bf16(false, a, false, b, (short)0, c, false, false);
}

// ---------- weight transpose+convert: W[K,N] f32 -> Wt[NPAD,KPAD] bf16 ----------
__global__ void wconv_kernel(const float* __restrict__ W, u16* __restrict__ Wt,
                             int K, int N, int KPAD, int NPAD) {
  int i = blockIdx.x * 256 + threadIdx.x;
  int total = NPAD * KPAD;
  if (i >= total) return;
  int n = i / KPAD, k = i % KPAD;
  float v = (n < N && k < K) ? W[(size_t)k * N + n] : 0.f;
  Wt[i] = f2bf(v);
}

// ---------- FPS: one block per point cloud ----------
__global__ __launch_bounds__(256)
void fps_kernel(const float* __restrict__ pos, int P, int S, float* __restrict__ q) {
  __shared__ float sp[4096 * 3];
  __shared__ int   s_idx[2048];
  __shared__ float red_d[256];
  __shared__ int   red_i[256];
  int b = blockIdx.x, tid = threadIdx.x;
  const float* pb = pos + (size_t)b * P * 3;
  for (int i = tid; i < P * 3; i += 256) sp[i] = pb[i];
  float mind[16];
#pragma unroll
  for (int t = 0; t < 16; ++t) mind[t] = 3.0e38f;
  int last = 0;
  __syncthreads();
  int PT = P / 256;
  for (int it = 0; it < S; ++it) {
    if (tid == 0) s_idx[it] = last;
    float lx = sp[last * 3], ly = sp[last * 3 + 1], lz = sp[last * 3 + 2];
    float bestd = -1.f; int besti = 0;
    for (int t = 0; t < PT; ++t) {
      int p = t * 256 + tid;
      float dx = sp[p * 3] - lx, dy = sp[p * 3 + 1] - ly, dz = sp[p * 3 + 2] - lz;
      float d2 = dx * dx + dy * dy + dz * dz;
      mind[t] = fminf(mind[t], d2);
      if (mind[t] > bestd) { bestd = mind[t]; besti = p; }
    }
    red_d[tid] = bestd; red_i[tid] = besti;
    __syncthreads();
    for (int off = 128; off > 0; off >>= 1) {
      if (tid < off && red_d[tid + off] > red_d[tid]) {
        red_d[tid] = red_d[tid + off]; red_i[tid] = red_i[tid + off];
      }
      __syncthreads();
    }
    last = red_i[0];
    __syncthreads();
  }
  for (int s = tid; s < S; s += 256) {
    int p = s_idx[s];
    q[((size_t)b * S + s) * 3 + 0] = sp[p * 3 + 0];
    q[((size_t)b * S + s) * 3 + 1] = sp[p * 3 + 1];
    q[((size_t)b * S + s) * 3 + 2] = sp[p * 3 + 2];
  }
}

// ---------- ball query: k=64 nearest within r ----------
__global__ void ballq_kernel(const float* __restrict__ q, const float* __restrict__ pts,
                             int* __restrict__ nidx, int NQ, int S, int P, float r2) {
  int i = blockIdx.x * 256 + threadIdx.x;
  if (i >= NQ) return;
  int b = i / S;
  const float* pb = pts + (size_t)b * P * 3;
  float qx = q[i * 3], qy = q[i * 3 + 1], qz = q[i * 3 + 2];
  float bd[64]; int bi[64];
  int cnt = 0, maxslot = 0; float maxd = -1.f;
  for (int p = 0; p < P; ++p) {
    float dx = pb[p * 3] - qx, dy = pb[p * 3 + 1] - qy, dz = pb[p * 3 + 2] - qz;
    float d2 = dx * dx + dy * dy + dz * dz;
    if (d2 > r2) continue;
    if (cnt < 64) {
      bd[cnt] = d2; bi[cnt] = p;
      if (d2 > maxd) { maxd = d2; maxslot = cnt; }
      cnt++;
    } else if (d2 < maxd) {
      bd[maxslot] = d2; bi[maxslot] = p;
      maxd = -1.f;
      for (int t = 0; t < 64; ++t) if (bd[t] > maxd) { maxd = bd[t]; maxslot = t; }
    }
  }
  int* o = nidx + (size_t)i * 64;
  for (int t = 0; t < 64; ++t) o[t] = (t < cnt) ? bi[t] : -1;
}

// ---------- fused SA layers (WMMA) ----------
template<int KP, int COUT>
__device__ __forceinline__ void sa_layer(const u16* Ain, const u16* Wt, const float* bias,
                                         u16* Hout, int lane, int wave) {
  const u16* arow = Ain + wave * 16 * KP;
#pragma unroll 1
  for (int nt = 0; nt < COUT / 16; ++nt) {
    v8f acc = {};
    for (int kc = 0; kc < KP; kc += 32) {
      v16bf a  = load_frag(arow + kc, KP, lane);
      v16bf bb = load_frag(Wt + (size_t)nt * 16 * KP + kc, KP, lane);
      acc = wmma_bf16(a, bb, acc);
    }
    int nl = lane & 15;
    float bv = bias[nt * 16 + nl];
#pragma unroll
    for (int v = 0; v < 8; ++v) {
      int m = wave * 16 + v + ((lane >> 4) << 3);
      float x = fmaxf(acc[v] + bv, 0.f);
      Hout[m * COUT + nt * 16 + nl] = f2bf(x);
    }
  }
}

template<int KP, int COUT>
__device__ __forceinline__ void sa_layer_max(const u16* Ain, const u16* Wt, const float* bias,
                                             float* s_part, const int* s_valid, int lane, int wave) {
  const u16* arow = Ain + wave * 16 * KP;
#pragma unroll 1
  for (int nt = 0; nt < COUT / 16; ++nt) {
    v8f acc = {};
    for (int kc = 0; kc < KP; kc += 32) {
      v16bf a  = load_frag(arow + kc, KP, lane);
      v16bf bb = load_frag(Wt + (size_t)nt * 16 * KP + kc, KP, lane);
      acc = wmma_bf16(a, bb, acc);
    }
    int nl = lane & 15;
    float bv = bias[nt * 16 + nl];
    float vmax = -3.0e38f;
#pragma unroll
    for (int v = 0; v < 8; ++v) {
      int row = wave * 16 + v + ((lane >> 4) << 3);
      float x = acc[v] + bv;
      if (s_valid[row]) vmax = fmaxf(vmax, x);
    }
    s_part[(wave * 2 + (lane >> 4)) * COUT + nt * 16 + nl] = vmax;
  }
}

// One block (4 waves) per query: gather 64 neighbor features, 3-layer MLP in LDS,
// max over neighbors, write [C3] bf16 row.
template<int KP0, int C1, int C2, int C3, int CXIN>
__global__ __launch_bounds__(128)
void sa_kernel(const float* __restrict__ pos, const u16* __restrict__ xin,
               const float* __restrict__ qpos, const int* __restrict__ nidx,
               const u16* w1, const float* b1, const u16* w2, const float* b2,
               const u16* w3, const float* b3, u16* __restrict__ xout, int P, int S) {
  __shared__ u16   s_feat[64 * KP0];
  __shared__ u16   s_h1[64 * C1];
  __shared__ u16   s_h2[64 * C2];
  __shared__ float s_part[8 * C3];
  __shared__ int   s_valid[64];
  int bs = blockIdx.x;
  int b = bs / S;
  int tid = threadIdx.x, lane = tid & 31, wave = tid >> 5;
  float qx = qpos[(size_t)bs * 3], qy = qpos[(size_t)bs * 3 + 1], qz = qpos[(size_t)bs * 3 + 2];
  if (tid < 64) {
    int j = tid;
    int nb = nidx[(size_t)bs * 64 + j];
    int valid = nb >= 0;
    s_valid[j] = valid;
    int nc = valid ? nb : 0;
    const float* pj = &pos[((size_t)b * P + nc) * 3];
    float rx = pj[0] - qx, ry = pj[1] - qy, rz = pj[2] - qz;
    u16* row = &s_feat[j * KP0];
    int c = 0;
    if (CXIN > 0) {
      const u16* xr = &xin[((size_t)b * P + nc) * CXIN];
      for (; c < CXIN; ++c) row[c] = xr[c];
    }
    row[c++] = f2bf(rx); row[c++] = f2bf(ry); row[c++] = f2bf(rz);
    for (; c < KP0; ++c) row[c] = 0;
  }
  __syncthreads();
  sa_layer<KP0, C1>(s_feat, w1, b1, s_h1, lane, wave);
  __syncthreads();
  sa_layer<C1, C2>(s_h1, w2, b2, s_h2, lane, wave);
  __syncthreads();
  sa_layer_max<C2, C3>(s_h2, w3, b3, s_part, s_valid, lane, wave);
  __syncthreads();
  for (int c = tid; c < C3; c += 128) {
    float m = s_part[c];
#pragma unroll
    for (int g = 1; g < 8; ++g) m = fmaxf(m, s_part[g * C3 + c]);
    xout[(size_t)bs * C3 + c] = f2bf(m);
  }
}

// ---------- generic WMMA GEMM: out = act(A[M,KPAD]bf16 @ Wt[N,KPAD]^T + bias) ----------
__global__ __launch_bounds__(128)
void gemm_kernel(const u16* __restrict__ A, const u16* __restrict__ Wt,
                 const float* __restrict__ bias, void* __restrict__ out,
                 int KPAD, int N, int biasN, int relu, int out_f32) {
  __shared__ u16 sA[64 * 32];
  int mbase = blockIdx.x * 64;
  int nbase = blockIdx.y * 64;
  int tid = threadIdx.x, lane = tid & 31, wave = tid >> 5;
  v8f acc[4] = {};
  for (int kc = 0; kc < KPAD; kc += 32) {
    __syncthreads();
    for (int i = tid; i < 64 * 16; i += 128) {  // 1024 dwords
      int row = i >> 4, kk = (i & 15) << 1;
      *reinterpret_cast<uint32_t*>(&sA[row * 32 + kk]) =
          *reinterpret_cast<const uint32_t*>(&A[(size_t)(mbase + row) * KPAD + kc + kk]);
    }
    __syncthreads();
    v16bf afrag = load_frag(&sA[wave * 16 * 32], 32, lane);
#pragma unroll
    for (int nt = 0; nt < 4; ++nt) {
      v16bf bfrag = load_frag(&Wt[(size_t)(nbase + nt * 16) * KPAD + kc], KPAD, lane);
      acc[nt] = wmma_bf16(afrag, bfrag, acc[nt]);
    }
  }
#pragma unroll
  for (int nt = 0; nt < 4; ++nt) {
    int n = nbase + nt * 16 + (lane & 15);
    float bv = (n < biasN) ? bias[n] : 0.f;
#pragma unroll
    for (int v = 0; v < 8; ++v) {
      int m = mbase + wave * 16 + v + ((lane >> 4) << 3);
      float x = acc[nt][v] + bv;
      if (relu) x = fmaxf(x, 0.f);
      if (out_f32) ((float*)out)[(size_t)m * N + n] = x;
      else         ((u16*)out)[(size_t)m * N + n] = f2bf(x);
    }
  }
}

// ---------- small scalar kernels ----------
__global__ void conv_relu_kernel(const float* __restrict__ in, const float* __restrict__ w,
                                 const float* __restrict__ bias, float* __restrict__ out,
                                 int B, int Cin, int Hin, int Win, int Cout,
                                 int KH, int KW, int stride, int Hout, int Wout) {
  int i = blockIdx.x * 256 + threadIdx.x;
  int total = B * Cout * Hout * Wout;
  if (i >= total) return;
  int wo = i % Wout, ho = (i / Wout) % Hout, co = (i / (Wout * Hout)) % Cout, b = i / (Wout * Hout * Cout);
  float acc = bias[co];
  for (int ci = 0; ci < Cin; ++ci)
    for (int kh = 0; kh < KH; ++kh)
      for (int kw = 0; kw < KW; ++kw)
        acc += in[((size_t)(b * Cin + ci) * Hin + ho * stride + kh) * Win + wo * stride + kw] *
               w[((size_t)(co * Cin + ci) * KH + kh) * KW + kw];
  out[i] = fmaxf(acc, 0.f);
}

__global__ void gsa_feats_kernel(const u16* __restrict__ x2, const float* __restrict__ q2,
                                 u16* __restrict__ Ag) {
  int i = blockIdx.x * 256 + threadIdx.x;
  if (i >= 4096 * 288) return;
  int r = i / 288, c = i % 288;
  u16 v;
  if (c < 256)      v = x2[(size_t)r * 256 + c];
  else if (c < 259) v = f2bf(q2[(size_t)r * 3 + (c - 256)]);
  else              v = 0;
  Ag[i] = v;
}

__global__ void zero16_kernel(u16* p, int n) {
  int i = blockIdx.x * 256 + threadIdx.x;
  if (i < n) p[i] = 0;
}

__global__ void cflat_kernel(const float* __restrict__ c3, u16* __restrict__ feat) {
  int i = blockIdx.x * 256 + threadIdx.x;
  if (i >= 8 * 960) return;
  int b = i / 960, c = i % 960;
  feat[(size_t)b * 1984 + c] = f2bf(c3[i]);
}

__global__ void pnmax_kernel(const float* __restrict__ H3, u16* __restrict__ feat) {
  int i = blockIdx.x * 256 + threadIdx.x;
  if (i >= 8 * 1024) return;
  int b = i >> 10, c = i & 1023;
  float m = -3.0e38f;
  for (int s = 0; s < 512; ++s) m = fmaxf(m, H3[((size_t)(b * 512 + s)) * 1024 + c]);
  feat[(size_t)b * 1984 + 960 + c] = f2bf(m);
}

__global__ void copyout_kernel(const float* __restrict__ Hfc2, float* __restrict__ out) {
  int i = blockIdx.x * 64 + threadIdx.x;
  if (i >= 160) return;
  int b = i / 20, c = i % 20;
  out[i] = Hfc2[b * 64 + c];
}

// ---------- host launch ----------
static inline size_t al256(size_t x) { return (x + 255) & ~(size_t)255; }
static inline int cdiv(int a, int b) { return (a + b - 1) / b; }

extern "C" void kernel_launch(void* const* d_in, const int* in_sizes, int n_in,
                              void* d_out, int out_size, void* d_ws, size_t ws_size,
                              hipStream_t stream) {
  (void)in_sizes; (void)n_in; (void)out_size; (void)ws_size;
  const float* img = (const float*)d_in[0];
  const float* pos = (const float*)d_in[1];
  const float* cw1 = (const float*)d_in[3];  const float* cb1 = (const float*)d_in[4];
  const float* cw2 = (const float*)d_in[5];  const float* cb2 = (const float*)d_in[6];
  const float* cw3 = (const float*)d_in[7];  const float* cb3 = (const float*)d_in[8];
  const float* s1w1 = (const float*)d_in[9];  const float* s1b1 = (const float*)d_in[10];
  const float* s1w2 = (const float*)d_in[11]; const float* s1b2 = (const float*)d_in[12];
  const float* s1w3 = (const float*)d_in[13]; const float* s1b3 = (const float*)d_in[14];
  const float* s2w1 = (const float*)d_in[15]; const float* s2b1 = (const float*)d_in[16];
  const float* s2w2 = (const float*)d_in[17]; const float* s2b2 = (const float*)d_in[18];
  const float* s2w3 = (const float*)d_in[19]; const float* s2b3 = (const float*)d_in[20];
  const float* s3w1 = (const float*)d_in[21]; const float* s3b1 = (const float*)d_in[22];
  const float* s3w2 = (const float*)d_in[23]; const float* s3b2 = (const float*)d_in[24];
  const float* s3w3 = (const float*)d_in[25]; const float* s3b3 = (const float*)d_in[26];
  const float* fc1w = (const float*)d_in[27]; const float* fc1b = (const float*)d_in[28];
  const float* fc2w = (const float*)d_in[29]; const float* fc2b = (const float*)d_in[30];

  char* w = (char*)d_ws;
  size_t off = 0;
  auto alloc = [&](size_t bytes) { void* p = w + off; off = al256(off + bytes); return p; };

  u16* wt_s1_1 = (u16*)alloc(64 * 32 * 2);
  u16* wt_s1_2 = (u16*)alloc(64 * 64 * 2);
  u16* wt_s1_3 = (u16*)alloc(128 * 64 * 2);
  u16* wt_s2_1 = (u16*)alloc(128 * 160 * 2);
  u16* wt_s2_2 = (u16*)alloc(128 * 128 * 2);
  u16* wt_s2_3 = (u16*)alloc(256 * 128 * 2);
  u16* wt_s3_1 = (u16*)alloc(256 * 288 * 2);
  u16* wt_s3_2 = (u16*)alloc((size_t)512 * 256 * 2);
  u16* wt_s3_3 = (u16*)alloc((size_t)1024 * 512 * 2);
  u16* wt_fc1  = (u16*)alloc((size_t)512 * 1984 * 2);
  u16* wt_fc2  = (u16*)alloc(64 * 512 * 2);
  float* q1   = (float*)alloc((size_t)8 * 2048 * 3 * 4);
  int*   nid1 = (int*)alloc((size_t)8 * 2048 * 64 * 4);
  u16*   x1   = (u16*)alloc((size_t)8 * 2048 * 128 * 2);
  float* q2   = (float*)alloc((size_t)8 * 512 * 3 * 4);
  int*   nid2 = (int*)alloc((size_t)8 * 512 * 64 * 4);
  u16*   x2   = (u16*)alloc((size_t)8 * 512 * 256 * 2);
  u16*   Ag   = (u16*)alloc((size_t)4096 * 288 * 2);
  u16*   Hg1  = (u16*)alloc((size_t)4096 * 256 * 2);
  u16*   Hg2  = (u16*)alloc((size_t)4096 * 512 * 2);
  float* Hg3  = (float*)alloc((size_t)4096 * 1024 * 4);
  float* c1   = (float*)alloc((size_t)8 * 32 * 15 * 11 * 4);
  float* c2   = (float*)alloc((size_t)8 * 64 * 7 * 5 * 4);
  float* c3   = (float*)alloc((size_t)8 * 64 * 5 * 3 * 4);
  u16*   featA = (u16*)alloc((size_t)64 * 1984 * 2);
  u16*   Hfc1  = (u16*)alloc((size_t)64 * 512 * 2);
  float* Hfc2  = (float*)alloc((size_t)64 * 64 * 4);

  auto wconv = [&](const float* W, u16* Wt, int K, int N, int KPAD, int NPAD) {
    int total = NPAD * KPAD;
    wconv_kernel<<<cdiv(total, 256), 256, 0, stream>>>(W, Wt, K, N, KPAD, NPAD);
  };
  wconv(s1w1, wt_s1_1, 3, 64, 32, 64);
  wconv(s1w2, wt_s1_2, 64, 64, 64, 64);
  wconv(s1w3, wt_s1_3, 64, 128, 64, 128);
  wconv(s2w1, wt_s2_1, 131, 128, 160, 128);
  wconv(s2w2, wt_s2_2, 128, 128, 128, 128);
  wconv(s2w3, wt_s2_3, 128, 256, 128, 256);
  wconv(s3w1, wt_s3_1, 259, 256, 288, 256);
  wconv(s3w2, wt_s3_2, 256, 512, 256, 512);
  wconv(s3w3, wt_s3_3, 512, 1024, 512, 1024);
  wconv(fc1w, wt_fc1, 1984, 512, 1984, 512);
  wconv(fc2w, wt_fc2, 512, 20, 512, 64);

  // CNN
  conv_relu_kernel<<<cdiv(8 * 32 * 15 * 11, 256), 256, 0, stream>>>(
      img, cw1, cb1, c1, 8, 1, 64, 48, 32, 8, 8, 4, 15, 11);
  conv_relu_kernel<<<cdiv(8 * 64 * 7 * 5, 256), 256, 0, stream>>>(
      c1, cw2, cb2, c2, 8, 32, 15, 11, 64, 3, 3, 2, 7, 5);
  conv_relu_kernel<<<cdiv(8 * 64 * 5 * 3, 256), 256, 0, stream>>>(
      c2, cw3, cb3, c3, 8, 64, 7, 5, 64, 3, 3, 1, 5, 3);

  // SA1: FPS 4096->2048, ball r=0.2 k=64, MLP 3->64->64->128, max
  fps_kernel<<<8, 256, 0, stream>>>(pos, 4096, 2048, q1);
  ballq_kernel<<<cdiv(8 * 2048, 256), 256, 0, stream>>>(q1, pos, nid1, 8 * 2048, 2048, 4096, 0.04f);
  sa_kernel<32, 64, 64, 128, 0><<<8 * 2048, 128, 0, stream>>>(
      pos, (const u16*)nullptr, q1, nid1,
      wt_s1_1, s1b1, wt_s1_2, s1b2, wt_s1_3, s1b3, x1, 4096, 2048);

  // SA2: FPS 2048->512, ball r=0.4 k=64, MLP 131->128->128->256, max
  fps_kernel<<<8, 256, 0, stream>>>(q1, 2048, 512, q2);
  ballq_kernel<<<cdiv(8 * 512, 256), 256, 0, stream>>>(q2, q1, nid2, 8 * 512, 512, 2048, 0.16f);
  sa_kernel<160, 128, 128, 256, 128><<<8 * 512, 128, 0, stream>>>(
      q1, x1, q2, nid2,
      wt_s2_1, s2b1, wt_s2_2, s2b2, wt_s2_3, s2b3, x2, 2048, 512);

  // Global SA: [4096,259] -> 256 -> 512 -> 1024, then max over 512 pts/batch
  gsa_feats_kernel<<<cdiv(4096 * 288, 256), 256, 0, stream>>>(x2, q2, Ag);
  gemm_kernel<<<dim3(64, 4), 128, 0, stream>>>(Ag, wt_s3_1, s3b1, Hg1, 288, 256, 256, 1, 0);
  gemm_kernel<<<dim3(64, 8), 128, 0, stream>>>(Hg1, wt_s3_2, s3b2, Hg2, 256, 512, 512, 1, 0);
  gemm_kernel<<<dim3(64, 16), 128, 0, stream>>>(Hg2, wt_s3_3, s3b3, Hg3, 512, 1024, 1024, 0, 1);

  // concat [cnn(960) | pnmax(1024)] into padded 64x1984 bf16 A
  zero16_kernel<<<cdiv(64 * 1984, 256), 256, 0, stream>>>(featA, 64 * 1984);
  cflat_kernel<<<cdiv(8 * 960, 256), 256, 0, stream>>>(c3, featA);
  pnmax_kernel<<<cdiv(8 * 1024, 256), 256, 0, stream>>>(Hg3, featA);

  // FC head
  gemm_kernel<<<dim3(1, 8), 128, 0, stream>>>(featA, wt_fc1, fc1b, Hfc1, 1984, 512, 512, 1, 0);
  gemm_kernel<<<dim3(1, 1), 128, 0, stream>>>(Hfc1, wt_fc2, fc2b, Hfc2, 512, 64, 20, 0, 1);
  copyout_kernel<<<cdiv(160, 64), 64, 0, stream>>>(Hfc2, (float*)d_out);
}
